// FlowPointContrast_65395172049369
// MI455X (gfx1250) — matrix operand order repaired and checked by reading, not compiled, not run on hardware
//
#include <hip/hip_runtime.h>
#include <hip/hip_bf16.h>

// ---------------- problem constants ----------------
#define BB 4
#define NN 4096
#define DD 64
#define NT (NN / 16)              // 256 column tiles / row tiles per batch
#define L2E 1.4426950408889634f   // log2(e)

typedef __attribute__((ext_vector_type(16))) __bf16 v16bf;
typedef __attribute__((ext_vector_type(8)))  __bf16 v8bf;
typedef __attribute__((ext_vector_type(8)))  float  v8f;
typedef __attribute__((ext_vector_type(2)))  float  v2f;

#if defined(__has_builtin)
#if __has_builtin(__builtin_amdgcn_wmma_f32_16x16x4_f32)
#define HAVE_WMMA4 1
#endif
#endif

static __device__ __forceinline__ unsigned short f2bf(float x) {
  union { float f; unsigned u; } v; v.f = x;
  unsigned u = v.u;
  unsigned r = u + 0x7FFFu + ((u >> 16) & 1u);   // round-to-nearest-even
  return (unsigned short)(r >> 16);
}

static __device__ __forceinline__ float fast_ln1p(float x) {
  return __builtin_amdgcn_logf(1.0f + x) * 0.6931471805599453f;
}

// ---------------------------------------------------------------------------
// Kernel 1: normalize features -> bf16; pack per-point attributes.
// attr layout per point (16 floats):
//   [0..3]  flow.x, flow.y, flow.z, 0
//   [4..7]  cx, cy, cz, 0            (c = color/255)
//   [8..11] px, py, pz, 0
//   [12..15]|flow|, |c|^2, |p|^2, 0
// ---------------------------------------------------------------------------
__global__ void prep_kernel(const float* __restrict__ flow,
                            const float* __restrict__ color,
                            const float* __restrict__ coord,
                            const float* __restrict__ feats,
                            unsigned short* __restrict__ fnb,
                            float* __restrict__ attr) {
  int idx = blockIdx.x * blockDim.x + threadIdx.x;
  if (idx >= BB * NN) return;

  const float* f = feats + (size_t)idx * DD;
  float v[DD];
  float ss = 0.0f;
#pragma unroll
  for (int i = 0; i < DD; ++i) { v[i] = f[i]; ss += v[i] * v[i]; }
  float inv = __builtin_amdgcn_rcpf(__builtin_amdgcn_sqrtf(ss) + 1e-7f);
  unsigned short* o = fnb + (size_t)idx * DD;
#pragma unroll
  for (int i = 0; i < DD; ++i) o[i] = f2bf(v[i] * inv);

  float fx = flow[idx * 3 + 0], fy = flow[idx * 3 + 1], fz = flow[idx * 3 + 2];
  float fn = __builtin_amdgcn_sqrtf(fx * fx + fy * fy + fz * fz);
  const float s255 = 1.0f / 255.0f;
  float cx = color[idx * 3 + 0] * s255, cy = color[idx * 3 + 1] * s255, cz = color[idx * 3 + 2] * s255;
  float c2 = cx * cx + cy * cy + cz * cz;
  float px = coord[idx * 3 + 0], py = coord[idx * 3 + 1], pz = coord[idx * 3 + 2];
  float p2 = px * px + py * py + pz * pz;

  float* a = attr + (size_t)idx * 16;
  a[0] = fx; a[1] = fy; a[2] = fz; a[3] = 0.0f;
  a[4] = cx; a[5] = cy; a[6] = cz; a[7] = 0.0f;
  a[8] = px; a[9] = py; a[10] = pz; a[11] = 0.0f;
  a[12] = fn; a[13] = c2; a[14] = p2; a[15] = 0.0f;
}

// ---------------------------------------------------------------------------
// Kernel 2: fused WMMA tile + contrastive gating + reduction.
// One block = one (batch, 16-row tile). 8 waves split 256 column tiles.
// ---------------------------------------------------------------------------
__global__ __launch_bounds__(256, 1)
void main_kernel(const unsigned short* __restrict__ fnb,
                 const float* __restrict__ attr,
                 float* __restrict__ partial) {
  const int b    = blockIdx.x >> 8;          // NT == 256
  const int n0   = (blockIdx.x & (NT - 1)) << 4;
  const int tid  = threadIdx.x;
  const int w    = tid >> 5;
  const int lane = tid & 31;
  const int hi   = lane >> 4;                // which half of the wave
  const int lc   = lane & 15;

  __shared__ float wavebuf[8][16][6];        // per-wave per-row partial sums
  __shared__ float rowloss[16];

  // ---- A fragments (constant over column loop) ----
  const size_t rowbase = (size_t)b * NN + n0 + lc;
  const unsigned short* arow = fnb + rowbase * DD;
  v16bf afrag[2];
#pragma unroll
  for (int ks = 0; ks < 2; ++ks) {
    v8bf h0 = *(const v8bf*)(arow + 32 * ks + 16 * hi);       // K = 16*hi .. +7
    v8bf h1 = *(const v8bf*)(arow + 32 * ks + 16 * hi + 8);   // K = 16*hi+8 .. +7
#pragma unroll
    for (int e = 0; e < 8; ++e) { afrag[ks][e] = h0[e]; afrag[ks][8 + e] = h1[e]; }
  }

#if HAVE_WMMA4
  // f32 16x16x4 A fragments: lane holds components {2*hi, 2*hi+1} of the 4-vec
  const float* arow_attr = attr + rowbase * 16;
  v2f a2[3];
#pragma unroll
  for (int s = 0; s < 3; ++s) {
    a2[s][0] = arow_attr[4 * s + 2 * hi];
    a2[s][1] = arow_attr[4 * s + 2 * hi + 1];
  }
#else
  float rvec[3][8][3];
#pragma unroll
  for (int s = 0; s < 3; ++s)
#pragma unroll
    for (int r = 0; r < 8; ++r)
#pragma unroll
      for (int k = 0; k < 3; ++k)
        rvec[s][r][k] = attr[((size_t)b * NN + n0 + 8 * hi + r) * 16 + 4 * s + k];
#endif

  // ---- row scalars for the 8 rows this lane's D elements belong to ----
  float rfn[8], rc2[8], rp2[8];
#pragma unroll
  for (int r = 0; r < 8; ++r) {
    const float* sc = attr + ((size_t)b * NN + n0 + 8 * hi + r) * 16 + 12;
    rfn[r] = sc[0]; rc2[r] = sc[1]; rp2[r] = sc[2];
  }

  // persistent per-row partial sums: [2*s] = s_plus, [2*s+1] = s_minus
  float acc[6][8];
#pragma unroll
  for (int s = 0; s < 6; ++s)
#pragma unroll
    for (int r = 0; r < 8; ++r) acc[s][r] = 0.0f;

  const float taus[3] = {0.8f, 0.7f, 0.5f};

  // ---- column-tile loop ----
  for (int t = w; t < NT; t += 8) {
    const int m0 = t << 4;
    const size_t colbase = (size_t)b * NN + m0 + lc;
    const unsigned short* brow = fnb + colbase * DD;

    // prefetch next tile's B operand into cache
    if (t + 8 < NT) {
      const unsigned short* nbrow = fnb + (colbase + 128) * DD;  // +8 tiles * 16 rows
      __builtin_prefetch(nbrow, 0, 3);
    }

    // bf16 B fragments: contiguous 16 bf16 at K-offset 16*hi within each K-step
    v16bf bfrag0 = *(const v16bf*)(brow + 16 * hi);
    v16bf bfrag1 = *(const v16bf*)(brow + 32 + 16 * hi);

    v8f cf = {0.f, 0.f, 0.f, 0.f, 0.f, 0.f, 0.f, 0.f};
    cf = __builtin_amdgcn_wmma_f32_16x16x32_bf16(false, afrag[0], false, bfrag0,
                                                 (short)0, cf, false, false);
    cf = __builtin_amdgcn_wmma_f32_16x16x32_bf16(false, afrag[1], false, bfrag1,
                                                 (short)0, cf, false, false);

    const float* cattr = attr + colbase * 16;
#if HAVE_WMMA4
    v8f dsim[3];
#pragma unroll
    for (int s = 0; s < 3; ++s) {
      v2f b2;
      b2[0] = cattr[4 * s + 2 * hi];
      b2[1] = cattr[4 * s + 2 * hi + 1];
      v8f z = {0.f, 0.f, 0.f, 0.f, 0.f, 0.f, 0.f, 0.f};
      dsim[s] = __builtin_amdgcn_wmma_f32_16x16x4_f32(false, a2[s], false, b2,
                                                      (short)0, z, false, false);
    }
#else
    float cfv[3][3];
#pragma unroll
    for (int s = 0; s < 3; ++s)
#pragma unroll
      for (int k = 0; k < 3; ++k) cfv[s][k] = cattr[4 * s + k];
#endif
    const float cfn = cattr[12], cc2 = cattr[13], cp2 = cattr[14];

#pragma unroll
    for (int r = 0; r < 8; ++r) {
      float fs = cf[r];
      float d0, d1, d2;
#if HAVE_WMMA4
      d0 = dsim[0][r]; d1 = dsim[1][r]; d2 = dsim[2][r];
#else
      d0 = rvec[0][r][0] * cfv[0][0] + rvec[0][r][1] * cfv[0][1] + rvec[0][r][2] * cfv[0][2];
      d1 = rvec[1][r][0] * cfv[1][0] + rvec[1][r][1] * cfv[1][1] + rvec[1][r][2] * cfv[1][2];
      d2 = rvec[2][r][0] * cfv[2][0] + rvec[2][r][1] * cfv[2][1] + rvec[2][r][2] * cfv[2][2];
#endif
      float sflow = d0 * __builtin_amdgcn_rcpf(rfn[r] * cfn + 1e-8f);
      float d2c = fmaxf(rc2[r] + cc2 - 2.0f * d1, 0.0f);
      float scol = 1.0f - __builtin_amdgcn_sqrtf(d2c) * 0.5773502691896258f;  // 1/sqrt(3)
      float d2p = fmaxf(rp2[r] + cp2 - 2.0f * d2, 0.0f);
      float sprox = __builtin_amdgcn_exp2f(-72.13475204444817f *             // 50*log2(e)
                                           __builtin_amdgcn_sqrtf(d2p));
      float sims[3] = {sflow, scol, sprox};
#pragma unroll
      for (int s = 0; s < 3; ++s) {
        // g+ = sigmoid(GAMMA*(sim - tau)); g- = 1 - g+
        float gp = __builtin_amdgcn_rcpf(
            1.0f + __builtin_amdgcn_exp2f(7.213475204444817f * (taus[s] - sims[s])));
        float gm = 1.0f - gp;
        acc[2 * s + 0][r] += __builtin_amdgcn_exp2f(L2E * fs * gp);
        acc[2 * s + 1][r] += __builtin_amdgcn_exp2f(-L2E * fs * gm);
      }
    }
  }

  // ---- deterministic reduction across the 16 lanes sharing each row ----
#pragma unroll
  for (int s = 0; s < 6; ++s)
#pragma unroll
    for (int r = 0; r < 8; ++r) {
      float v = acc[s][r];
      v += __shfl_xor(v, 1, 32);
      v += __shfl_xor(v, 2, 32);
      v += __shfl_xor(v, 4, 32);
      v += __shfl_xor(v, 8, 32);
      acc[s][r] = v;
    }
  if (lc == 0) {
#pragma unroll
    for (int r = 0; r < 8; ++r)
#pragma unroll
      for (int s = 0; s < 6; ++s)
        wavebuf[w][8 * hi + r][s] = acc[s][r];
  }
  __syncthreads();

  if (tid < 16) {
    float L = 0.0f;
#pragma unroll
    for (int s = 0; s < 3; ++s) {
      float sp = 0.0f, sm = 0.0f;
#pragma unroll
      for (int ww = 0; ww < 8; ++ww) {
        sp += wavebuf[ww][tid][2 * s + 0];
        sm += wavebuf[ww][tid][2 * s + 1];
      }
      L += fast_ln1p(sp) + fast_ln1p(sm);   // ETA = NU = 1
    }
    rowloss[tid] = L;
  }
  __syncthreads();

  if (tid == 0) {
    float tsum = 0.0f;
#pragma unroll
    for (int i = 0; i < 16; ++i) tsum += rowloss[i];
    partial[blockIdx.x] = tsum;
  }
}

// ---------------------------------------------------------------------------
// Kernel 3: reduce 1024 block partials -> scalar loss.
// ---------------------------------------------------------------------------
__global__ void reduce_kernel(const float* __restrict__ partial, float* __restrict__ out) {
  __shared__ float sh[256];
  int tid = threadIdx.x;
  float s = partial[tid] + partial[tid + 256] + partial[tid + 512] + partial[tid + 768];
  sh[tid] = s;
  __syncthreads();
  for (int off = 128; off > 0; off >>= 1) {
    if (tid < off) sh[tid] += sh[tid + off];
    __syncthreads();
  }
  if (tid == 0) out[0] = -sh[0] / (float)(BB * NN);
}

// ---------------------------------------------------------------------------
extern "C" void kernel_launch(void* const* d_in, const int* in_sizes, int n_in,
                              void* d_out, int out_size, void* d_ws, size_t ws_size,
                              hipStream_t stream) {
  const float* flow  = (const float*)d_in[0];
  const float* color = (const float*)d_in[1];
  const float* coord = (const float*)d_in[2];
  const float* feats = (const float*)d_in[3];

  // workspace layout
  unsigned short* fnb = (unsigned short*)d_ws;                       // 2 MB bf16 fn
  size_t off_attr = (size_t)BB * NN * DD * sizeof(unsigned short);
  float* attr = (float*)((char*)d_ws + off_attr);                    // 1 MB attrs
  size_t off_part = off_attr + (size_t)BB * NN * 16 * sizeof(float);
  float* partial = (float*)((char*)d_ws + off_part);                 // 4 KB partials

  prep_kernel<<<(BB * NN + 255) / 256, 256, 0, stream>>>(flow, color, coord, feats, fnb, attr);
  main_kernel<<<BB * NT, 256, 0, stream>>>(fnb, attr, partial);
  reduce_kernel<<<1, 256, 0, stream>>>(partial, (float*)d_out);
}
